// Attention_65798898975517
// MI455X (gfx1250) — matrix operand order repaired and checked by reading, not compiled, not run on hardware
//
#include <hip/hip_runtime.h>
#include <hip/hip_bf16.h>

// ---------------- types / helpers ----------------
typedef __attribute__((ext_vector_type(16))) __bf16        v16bf;
typedef __attribute__((ext_vector_type(8)))  float         v8f;
typedef __attribute__((ext_vector_type(4)))  unsigned int  u32x4;
typedef __attribute__((ext_vector_type(4)))  float         f32x4;
typedef __attribute__((ext_vector_type(4)))  unsigned short u16x4;

static constexpr int Bq  = 4;
static constexpr int Tq  = 2048;
static constexpr int Dq  = 1024;
static constexpr int Hq  = 16;
static constexpr int DKq = 64;        // Dq / Hq
static constexpr int Mq  = Bq * Tq;   // 8192 rows

__device__ __forceinline__ unsigned f2bf_bits(float f) {
    unsigned u = __float_as_uint(f);
    u += 0x7fffu + ((u >> 16) & 1u);   // round-to-nearest-even
    return u >> 16;
}

// Assemble a 16-element bf16 fragment from two contiguous 16B chunks.
//   A (16x32, 16-bit): lane row = lane&15; elems 0..7 -> k = 8*half+0..7,
//                      elems 8..15 -> k = 16+8*half+0..7  (p0, p1)
//   B (32x16, 16-bit): lane col = lane&15; same k order as A (p0, p1)
__device__ __forceinline__ v16bf load_frag(const unsigned short* p0,
                                           const unsigned short* p1) {
    union { u32x4 q[2]; v16bf v; } u;
    u.q[0] = *(const u32x4*)p0;
    u.q[1] = *(const u32x4*)p1;
    return u.v;
}

__device__ __forceinline__ v8f wmma_bf16(v16bf a, v16bf b, v8f c) {
    return __builtin_amdgcn_wmma_f32_16x16x32_bf16(
        false, a, false, b, (short)0, c, false, false);
}

// Async global -> LDS copy, 16B per lane (ASYNCcnt-tracked, CDNA5).
// Same INST_OFFSET applies to both the LDS and global sides (ISA 10.7/15.18).
__device__ __forceinline__ void async_ld16(unsigned lds_off,
                                           unsigned long long gaddr) {
    asm volatile("global_load_async_to_lds_b128 %0, %1, off"
                 :: "v"(lds_off), "v"(gaddr) : "memory");
}
__device__ __forceinline__ void async_ld16_off16(unsigned lds_off,
                                                 unsigned long long gaddr) {
    asm volatile("global_load_async_to_lds_b128 %0, %1, off offset:16"
                 :: "v"(lds_off), "v"(gaddr) : "memory");
}

// ---------------- kernel 0: fp32 -> bf16 conversion ----------------
__global__ __launch_bounds__(256)
void cvt_kernel(const float* __restrict__ src, unsigned short* __restrict__ dst,
                int n4) {
    int i = blockIdx.x * blockDim.x + threadIdx.x;
    if (i >= n4) return;
    f32x4 f = *(const f32x4*)(src + (size_t)i * 4);
    u16x4 o;
    o[0] = (unsigned short)f2bf_bits(f[0]);
    o[1] = (unsigned short)f2bf_bits(f[1]);
    o[2] = (unsigned short)f2bf_bits(f[2]);
    o[3] = (unsigned short)f2bf_bits(f[3]);
    *(u16x4*)(dst + (size_t)i * 4) = o;
}

// ---------------- kernel 1: fused QKV projection ----------------
// y = x @ W^T + b (bf16 in/out). Wave -> 64x64 tile (16 WMMA / k-step).
// z=0: Q head-split [B,H,T,DK]; z=1: K head-split; z=2: V transposed [B,H,DK,T].
__global__ __launch_bounds__(128)
void qkv_proj_kernel(const unsigned short* __restrict__ xb,
                     const unsigned short* __restrict__ wqb,
                     const unsigned short* __restrict__ wkb,
                     const unsigned short* __restrict__ wvb,
                     const float* __restrict__ bq, const float* __restrict__ bk,
                     const float* __restrict__ bv,
                     unsigned short* __restrict__ qws,
                     unsigned short* __restrict__ kws,
                     unsigned short* __restrict__ vws) {
    const unsigned short* W; const float* bias; unsigned short* out;
    if (blockIdx.z == 0)      { W = wqb; bias = bq; out = qws; }
    else if (blockIdx.z == 1) { W = wkb; bias = bk; out = kws; }
    else                      { W = wvb; bias = bv; out = vws; }

    const int lane = threadIdx.x & 31;
    const int wave = threadIdx.x >> 5;
    const int half = lane >> 4;
    const int lm   = lane & 15;
    const int m0   = (blockIdx.y * 4 + wave) * 64;
    const int n0   = blockIdx.x * 64;

    v8f acc[4][4] = {};
    for (int k0 = 0; k0 < Dq; k0 += 32) {
        v16bf a[4];
#pragma unroll
        for (int sm = 0; sm < 4; ++sm) {
            const unsigned short* ar = xb + (size_t)(m0 + 16 * sm + lm) * Dq + k0;
            a[sm] = load_frag(ar + 8 * half, ar + 16 + 8 * half);
        }
#pragma unroll
        for (int i = 0; i < 4; ++i) {
            const unsigned short* br = W + (size_t)(n0 + i * 16 + lm) * Dq + k0
                                         + 8 * half;
            v16bf b = load_frag(br, br + 16);
#pragma unroll
            for (int sm = 0; sm < 4; ++sm)
                acc[sm][i] = wmma_bf16(a[sm], b, acc[sm][i]);
        }
    }
#pragma unroll
    for (int i = 0; i < 4; ++i) {
        const int ng = n0 + i * 16 + lm;
        const int h  = ng >> 6;
        const int dk = ng & 63;
        const float bv_ = bias[ng];
#pragma unroll
        for (int sm = 0; sm < 4; ++sm)
#pragma unroll
            for (int r = 0; r < 8; ++r) {
                const int mg = m0 + 16 * sm + r + 8 * half;
                const int bb = mg >> 11;
                const int t  = mg & (Tq - 1);
                const unsigned short v =
                    (unsigned short)f2bf_bits(acc[sm][i][r] + bv_);
                if (blockIdx.z == 2)
                    out[(((size_t)(bb * Hq + h) * DKq) + dk) * Tq + t] = v;
                else
                    out[(((size_t)(bb * Hq + h) * Tq) + t) * DKq + dk] = v;
            }
    }
}

// ---------------- kernel 2: causal flash attention ----------------
// Block = 128 threads (4 waves) -> 64 consecutive queries of one (b,h).
// K/V 32-key tiles staged via async global->LDS (ASYNCcnt), padded LDS rows.
__global__ __launch_bounds__(128)
void attn_kernel(const unsigned short* __restrict__ qws,
                 const unsigned short* __restrict__ kws,
                 const unsigned short* __restrict__ vws,  // [B,H,DK,T]
                 unsigned short* __restrict__ ows) {
    __shared__ unsigned short kl[32][72];        // 32 keys x 64 dims (pad 72)
    __shared__ unsigned short vl[64][40];        // 64 dims x 32 keys (pad 40)
    __shared__ unsigned short plds[4][16][32];   // per-wave P staging

    const int tid  = threadIdx.x;
    const int lane = tid & 31;
    const int wave = tid >> 5;
    const int half = lane >> 4;
    const int lm   = lane & 15;

    const int qb = blockIdx.x & 31;              // T/64 = 32 query blocks
    const int bh = blockIdx.x >> 5;              // 0..63
    const int q0 = qb * 64 + wave * 16;          // this wave's query tile
    const unsigned short* Q  = qws + (size_t)bh * Tq * DKq;
    const unsigned short* K  = kws + (size_t)bh * Tq * DKq;
    const unsigned short* Vt = vws + (size_t)bh * DKq * Tq;

    // Q fragments: rows q0..q0+15, depth halves of DK=64
    v16bf aq[2];
#pragma unroll
    for (int f = 0; f < 2; ++f) {
        const unsigned short* qr = Q + (size_t)(q0 + lm) * DKq + 32 * f;
        aq[f] = load_frag(qr + 8 * half, qr + 16 + 8 * half);
    }

    v8f o[4] = {};
    float Mrow[8], Lrow[8];
#pragma unroll
    for (int r = 0; r < 8; ++r) { Mrow[r] = -1e30f; Lrow[r] = 0.0f; }

    // staging roles (block-wide): 32B per thread for K and for V
    const int krow = tid >> 2, kch = (tid & 3) * 16;   // K: 32 rows x 4 chunks
    const int vdk  = tid >> 1, vch = (tid & 1) * 16;   // V: 64 rows x 2 chunks
    const unsigned koff = (unsigned)(uintptr_t)&kl[krow][kch];
    const unsigned voff = (unsigned)(uintptr_t)&vl[vdk][vch];

    const int nkt = 2 * qb + 2;                  // key tiles covering qb*64+63
    for (int jt = 0; jt < nkt; ++jt) {
        const int k0 = jt * 32;

        // ---- async stage of K/V tile into LDS (no VGPR bounce) ----
        {
            const unsigned long long kga = (unsigned long long)(uintptr_t)
                (K + (size_t)(k0 + krow) * DKq + kch);
            async_ld16(koff, kga);
            async_ld16_off16(koff, kga);
            const unsigned long long vga = (unsigned long long)(uintptr_t)
                (Vt + (size_t)vdk * Tq + k0 + vch);
            async_ld16(voff, vga);
            async_ld16_off16(voff, vga);
        }
        if (jt + 1 < nkt) {  // prefetch next tile (global_prefetch_b8)
            __builtin_prefetch(K + (size_t)(k0 + 32 + krow) * DKq + kch, 0, 0);
            __builtin_prefetch(Vt + (size_t)vdk * Tq + k0 + 32 + vch, 0, 0);
        }
        asm volatile("s_wait_asynccnt 0" ::: "memory");  // our transfers landed
        __syncthreads();                                 // all waves' landed

        if (k0 <= q0 + 15) {   // wave-uniform causal range check
            // ---- S = (Q K^T) * scale (two 16x16 column tiles) ----
            v8f s[2];
#pragma unroll
            for (int sidx = 0; sidx < 2; ++sidx) {
                const unsigned short* kr = &kl[16 * sidx + lm][8 * half];
                v16bf b0 = load_frag(kr,      kr + 16);
                v16bf b1 = load_frag(kr + 32, kr + 48);
                v8f t = {};
                t = wmma_bf16(aq[0], b0, t);
                t = wmma_bf16(aq[1], b1, t);
                s[sidx] = t;
            }
#pragma unroll
            for (int sidx = 0; sidx < 2; ++sidx) {
                const int keyg = k0 + 16 * sidx + lm;
#pragma unroll
                for (int r = 0; r < 8; ++r) {
                    const int qg = q0 + r + 8 * half;
                    const float v = s[sidx][r] * 0.125f;   // 1/sqrt(64)
                    s[sidx][r] = (keyg > qg) ? -1e30f : v;
                }
            }

            // ---- online softmax (cross-lane reductions, wave32) ----
#pragma unroll
            for (int r = 0; r < 8; ++r) {
                float mx = fmaxf(s[0][r], s[1][r]);
#pragma unroll
                for (int m = 1; m < 16; m <<= 1)
                    mx = fmaxf(mx, __shfl_xor(mx, m, 32));
                const float nm = fmaxf(Mrow[r], mx);
                const float sc = __expf(Mrow[r] - nm);
                const float p0 = __expf(s[0][r] - nm);
                const float p1 = __expf(s[1][r] - nm);
                float rs = p0 + p1;
#pragma unroll
                for (int m = 1; m < 16; m <<= 1)
                    rs += __shfl_xor(rs, m, 32);
                Lrow[r] = Lrow[r] * sc + rs;
                Mrow[r] = nm;
#pragma unroll
                for (int i = 0; i < 4; ++i) o[i][r] *= sc;
                const int mrow = r + 8 * half;
                plds[wave][mrow][lm]      = (unsigned short)f2bf_bits(p0);
                plds[wave][mrow][16 + lm] = (unsigned short)f2bf_bits(p1);
            }
            asm volatile("s_wait_dscnt 0" ::: "memory");  // same-wave DS RAW

            // ---- O += P @ V ----
            v16bf ap = load_frag(&plds[wave][lm][8 * half],
                                 &plds[wave][lm][16 + 8 * half]);
#pragma unroll
            for (int i = 0; i < 4; ++i) {
                const unsigned short* vr = &vl[i * 16 + lm][8 * half];
                v16bf bvf = load_frag(vr, vr + 16);
                o[i] = wmma_bf16(ap, bvf, o[i]);
            }
        }
        __syncthreads();
    }

    // ---- normalize, store bf16 O in [B,T,D] ----
    const int b_ = bh >> 4;
    const int h  = bh & 15;
#pragma unroll
    for (int i = 0; i < 4; ++i)
#pragma unroll
        for (int r = 0; r < 8; ++r) {
            const int t = q0 + r + 8 * half;
            const float val = o[i][r] / Lrow[r];
            ows[((size_t)(b_ * Tq + t)) * Dq + h * DKq + i * 16 + lm] =
                (unsigned short)f2bf_bits(val);
        }
}

// ---------------- kernel 3: output projection ----------------
// out = O @ Wo^T + bo (fp32 out). Wave -> 64x64 tile.
__global__ __launch_bounds__(128)
void out_proj_kernel(const unsigned short* __restrict__ ows,
                     const unsigned short* __restrict__ wob,
                     const float* __restrict__ bo, float* __restrict__ out) {
    const int lane = threadIdx.x & 31;
    const int wave = threadIdx.x >> 5;
    const int half = lane >> 4;
    const int lm   = lane & 15;
    const int m0   = (blockIdx.y * 4 + wave) * 64;
    const int n0   = blockIdx.x * 64;

    v8f acc[4][4] = {};
    for (int k0 = 0; k0 < Dq; k0 += 32) {
        v16bf a[4];
#pragma unroll
        for (int sm = 0; sm < 4; ++sm) {
            const unsigned short* ar = ows + (size_t)(m0 + 16 * sm + lm) * Dq + k0;
            a[sm] = load_frag(ar + 8 * half, ar + 16 + 8 * half);
        }
#pragma unroll
        for (int i = 0; i < 4; ++i) {
            const unsigned short* br = wob + (size_t)(n0 + i * 16 + lm) * Dq + k0
                                           + 8 * half;
            v16bf b = load_frag(br, br + 16);
#pragma unroll
            for (int sm = 0; sm < 4; ++sm)
                acc[sm][i] = wmma_bf16(a[sm], b, acc[sm][i]);
        }
    }
#pragma unroll
    for (int i = 0; i < 4; ++i) {
        const int ng = n0 + i * 16 + lm;
        const float bv_ = bo[ng];
#pragma unroll
        for (int sm = 0; sm < 4; ++sm)
#pragma unroll
            for (int r = 0; r < 8; ++r) {
                const int mg = m0 + 16 * sm + r + 8 * half;
                out[(size_t)mg * Dq + ng] = acc[sm][i][r] + bv_;
            }
    }
}

// ---------------- launcher ----------------
extern "C" void kernel_launch(void* const* d_in, const int* in_sizes, int n_in,
                              void* d_out, int out_size, void* d_ws, size_t ws_size,
                              hipStream_t stream) {
    const float* x  = (const float*)d_in[0];
    const float* Wq = (const float*)d_in[1];
    const float* bq = (const float*)d_in[2];
    const float* Wk = (const float*)d_in[3];
    const float* bk = (const float*)d_in[4];
    const float* Wv = (const float*)d_in[5];
    const float* bv = (const float*)d_in[6];
    const float* Wo = (const float*)d_in[7];
    const float* bo = (const float*)d_in[8];

    const size_t MD = (size_t)Mq * Dq;   // 8M elements
    const size_t DD = (size_t)Dq * Dq;   // 1M elements
    unsigned short* base = (unsigned short*)d_ws;
    unsigned short* qws = base;               // 16 MB
    unsigned short* kws = qws + MD;           // 16 MB
    unsigned short* vws = kws + MD;           // 16 MB (transposed V)
    unsigned short* ows = vws + MD;           // 16 MB
    unsigned short* xb  = ows + MD;           // 16 MB
    unsigned short* wqb = xb  + MD;           // 2 MB
    unsigned short* wkb = wqb + DD;
    unsigned short* wvb = wkb + DD;
    unsigned short* wob = wvb + DD;           // total 88 MB

    // 0) fp32 -> bf16 pre-conversion (bandwidth-trivial, off the hot path)
    cvt_kernel<<<(int)(MD / 4 / 256), 256, 0, stream>>>(x,  xb,  (int)(MD / 4));
    cvt_kernel<<<(int)(DD / 4 / 256), 256, 0, stream>>>(Wq, wqb, (int)(DD / 4));
    cvt_kernel<<<(int)(DD / 4 / 256), 256, 0, stream>>>(Wk, wkb, (int)(DD / 4));
    cvt_kernel<<<(int)(DD / 4 / 256), 256, 0, stream>>>(Wv, wvb, (int)(DD / 4));
    cvt_kernel<<<(int)(DD / 4 / 256), 256, 0, stream>>>(Wo, wob, (int)(DD / 4));

    // 1) fused QKV projection (z selects Q/K/V; V stored transposed)
    qkv_proj_kernel<<<dim3(Dq / 64, Mq / 64 / 4, 3), 128, 0, stream>>>(
        xb, wqb, wkb, wvb, bq, bk, bv, qws, kws, vws);

    // 2) causal flash attention: 2048 blocks, each 64 queries of one (b,h)
    attn_kernel<<<dim3(2048), 128, 0, stream>>>(qws, kws, vws, ows);

    // 3) output projection to fp32
    out_proj_kernel<<<dim3(Dq / 64, Mq / 64 / 4), 128, 0, stream>>>(
        ows, wob, bo, (float*)d_out);
}
